// Starcoder2Attention_6665789243457
// MI455X (gfx1250) — matrix operand order repaired and checked by reading, not compiled, not run on hardware
//
#include <hip/hip_runtime.h>
#include <hip/hip_bf16.h>
#include <stdint.h>

// ---------------- problem constants (from reference) ----------------
#define H_HEADS   24
#define KV_HEADS  2
#define DHEAD     128
#define HIDDEN    (H_HEADS * DHEAD)                  // 3072
#define QKV_N     ((H_HEADS + 2 * KV_HEADS) * DHEAD) // 3584
#define GROUPS    (H_HEADS / KV_HEADS)               // 12

typedef __bf16 bf16_t;
typedef __attribute__((ext_vector_type(16))) __bf16 v16bf;
typedef __attribute__((ext_vector_type(8)))  __bf16 v8bf;
typedef __attribute__((ext_vector_type(8)))  float  v8f;

union BF16x16 { v16bf v; v8bf h[2]; };

// Load a 16-element bf16 fragment from two 16-byte chunks.
__device__ __forceinline__ v16bf ld16(const bf16_t* p0, const bf16_t* p1) {
  BF16x16 u;
  u.h[0] = *(const v8bf*)p0;
  u.h[1] = *(const v8bf*)p1;
  return u.v;
}

__device__ __forceinline__ bf16_t f2bf(float f) {
  unsigned u = __builtin_bit_cast(unsigned, f);
  unsigned r = u + 0x7FFFu + ((u >> 16) & 1u);   // round-to-nearest-even
  unsigned short s = (unsigned short)(r >> 16);
  return __builtin_bit_cast(bf16_t, s);
}

// ---------------- fp32 -> bf16 bulk convert ----------------
__global__ void sc2_cvt_bf16(const float* __restrict__ in,
                             bf16_t* __restrict__ out, long n4) {
  long i = (long)blockIdx.x * blockDim.x + threadIdx.x;
  if (i >= n4) return;
  long idx = i * 4;
  float4 f = *(const float4*)(in + idx);
  bf16_t o[4] = { f2bf(f.x), f2bf(f.y), f2bf(f.z), f2bf(f.w) };
  *(uint64_t*)(out + idx) = *(const uint64_t*)o;
}

// ---------------- bf16 WMMA GEMM with async A-tile staging ----------------
// C[M,N] = A[M,K] * W[N,K]^T + bias
// Block tile 64(M) x 128(N); 8 waves, each wave computes 64x16.
// A tile (64x32 bf16 per k-step) is DMA'd to LDS with
// global_load_async_to_lds_b128 (one 16B chunk per thread), double buffered.
// All A-fragment LDS reads issue before the WMMA burst (single DScnt wait),
// and the B fragment is software-pipelined one k-step ahead.
#define GM_WAVES 8
#define APAD 40                 // padded LDS row stride in bf16 (80B rows)
#define ATILE_B (64 * APAD * 2) // bytes per A buffer

__global__ void __launch_bounds__(GM_WAVES * 32)
sc2_gemm_bf16(const bf16_t* __restrict__ A, const bf16_t* __restrict__ Bw,
              const float* __restrict__ bias, float* __restrict__ C,
              int N, int Kd) {
  __shared__ bf16_t tileA[2][64 * APAD];

  const int tid  = threadIdx.x;
  const int lane = tid & 31;
  const int wave = tid >> 5;
  const int n15  = lane & 15;
  const int hi   = (lane >> 4) & 1;
  const int rowb = blockIdx.y * 64;
  const int col  = blockIdx.x * 128 + wave * 16 + n15;

  const int ka  = hi ? 8 : 0;    // A-fragment K phase (lanes 16-31: K=8..15,24..31)
  const int kbb = hi ? 16 : 0;   // unused; B fragment uses ka, same K order as A

  // Per-thread async-copy slot: 16 bytes of the 64x32 A tile.
  const int arow = tid >> 2;           // 0..63
  const int acol = (tid & 3) * 8;      // bf16 column within row (0,8,16,24)
  const bf16_t* gA = A + (long)(rowb + arow) * Kd + acol;
  // LDS offset of generic pointer = low 32 bits (aperture lives in [63:32]).
  const unsigned lbase = (unsigned)(size_t)&tileA[0][0];
  const unsigned loff  = (unsigned)(arow * (APAD * 2) + (tid & 3) * 16);

#define ISSUE_ASYNC(buf, k0)                                                   \
  do {                                                                         \
    unsigned _d = lbase + (unsigned)(buf) * (unsigned)ATILE_B + loff;          \
    unsigned long long _g = (unsigned long long)(gA + (k0));                   \
    asm volatile("global_load_async_to_lds_b128 %0, %1, off"                   \
                 :: "v"(_d), "v"(_g) : "memory");                              \
  } while (0)

  v8f acc[4] = {};
  ISSUE_ASYNC(0, 0);

  // Prime the B fragment for k0 = 0 (16 cols x 32 K, contiguous along K).
  const bf16_t* bcol = Bw + (long)col * Kd + ka;
  v16bf bcur = ld16(bcol, bcol + 16);

  for (int k0 = 0, i = 0; k0 < Kd; k0 += 32, ++i) {
    const int cur = i & 1;
    // Our chunks for buf[cur] have landed in LDS; barrier publishes the tile
    // and guarantees every wave finished reading buf[cur^1] last iteration.
    asm volatile("s_wait_asynccnt 0x0" ::: "memory");
    __syncthreads();
    if (k0 + 32 < Kd) ISSUE_ASYNC(cur ^ 1, k0 + 32);

    // Issue all LDS reads for the 4 A fragments up front -> one DScnt wait.
    const bf16_t* at = &tileA[cur][0];
    v16bf af[4];
#pragma unroll
    for (int t = 0; t < 4; ++t) {
      const bf16_t* ap = at + (t * 16 + n15) * APAD + ka;
      af[t] = ld16(ap, ap + 16);
    }
    // Prefetch next iteration's B fragment (branch-free clamped address);
    // its loadcnt wait lands a full k-step away from this WMMA burst.
    const int kn = (k0 + 32 < Kd) ? (k0 + 32) : 0;
    v16bf bnext = ld16(bcol + kn, bcol + kn + 16);

#pragma unroll
    for (int t = 0; t < 4; ++t)
      acc[t] = __builtin_amdgcn_wmma_f32_16x16x32_bf16(
          false, af[t], false, bcur, (short)0, acc[t], false, false);
    bcur = bnext;
  }
#undef ISSUE_ASYNC

  const float bv = bias ? bias[col] : 0.0f;
#pragma unroll
  for (int t = 0; t < 4; ++t)
#pragma unroll
    for (int v = 0; v < 8; ++v) {
      int r = rowb + t * 16 + v + hi * 8;   // C layout: row = vgpr + 8*half
      C[(long)r * N + col] = acc[t][v] + bv;
    }
}

// ---------------- RoPE + repack to bf16 (Q, K) and transposed V ----------------
__global__ void sc2_rope_pack(const float* __restrict__ qkv,
                              const float* __restrict__ cosb,
                              const float* __restrict__ sinb,
                              bf16_t* __restrict__ q_bf,
                              bf16_t* __restrict__ k_bf,
                              bf16_t* __restrict__ vt_bf,
                              const int* __restrict__ seq_len_p) {
  const int S   = seq_len_p[0];
  const int tok = blockIdx.x;
  for (int w = threadIdx.x; w < (H_HEADS + 2 * KV_HEADS) * 64; w += blockDim.x) {
    const int head = w >> 6;
    const int i    = w & 63;
    if (head < H_HEADS) {                       // Q heads: rotate + pack
      const float* src = qkv + (long)tok * QKV_N + head * DHEAD;
      float x1 = src[i], x2 = src[i + 64];
      float c = cosb[tok * 64 + i], s = sinb[tok * 64 + i];
      bf16_t* dst = q_bf + (long)tok * HIDDEN + head * DHEAD;
      dst[i]      = f2bf(x1 * c - x2 * s);
      dst[i + 64] = f2bf(x2 * c + x1 * s);
    } else if (head < H_HEADS + KV_HEADS) {     // K heads: rotate + pack
      const int kv = head - H_HEADS;
      const float* src = qkv + (long)tok * QKV_N + HIDDEN + kv * DHEAD;
      float x1 = src[i], x2 = src[i + 64];
      float c = cosb[tok * 64 + i], s = sinb[tok * 64 + i];
      bf16_t* dst = k_bf + (long)tok * (KV_HEADS * DHEAD) + kv * DHEAD;
      dst[i]      = f2bf(x1 * c - x2 * s);
      dst[i + 64] = f2bf(x2 * c + x1 * s);
    } else {                                    // V heads: transpose to [b,kv][D][S]
      const int kv = head - (H_HEADS + KV_HEADS);
      const float* src = qkv + (long)tok * QKV_N + HIDDEN + KV_HEADS * DHEAD + kv * DHEAD;
      const int b = tok / S, sp = tok - b * S;
      bf16_t* dst = vt_bf + ((long)(b * KV_HEADS + kv) * DHEAD) * S + sp;
      dst[(long)i * S]        = f2bf(src[i]);
      dst[(long)(i + 64) * S] = f2bf(src[i + 64]);
    }
  }
}

// ---------------- flash attention: one wave per 16-query tile ----------------
#define ATT_WAVES 4
#define PLDS 40   // row stride (bf16) for P tile: 80B rows -> conflict-free 16B reads

__global__ void __launch_bounds__(ATT_WAVES * 32)
sc2_attn_wmma(const bf16_t* __restrict__ q_bf,
              const bf16_t* __restrict__ k_bf,
              const bf16_t* __restrict__ vt_bf,
              bf16_t* __restrict__ o_bf,
              const int* __restrict__ seq_len_p,
              const int* __restrict__ window_p) {
  __shared__ bf16_t pl[ATT_WAVES][16 * PLDS];
  const int S = seq_len_p[0];
  const int W = window_p[0];
  const int lane = threadIdx.x & 31;
  const int wave = threadIdx.x >> 5;
  const int n15  = lane & 15;
  const int hi   = (lane >> 4) & 1;
  const int h    = blockIdx.y;
  const int kvh  = h / GROUPS;
  const int tok0 = (blockIdx.x * ATT_WAVES + wave) * 16;
  const int b     = tok0 / S;
  const int qbase = tok0 - b * S;
  const float scale = 0.08838834764831845f;     // D^-0.5

  const int ka  = hi ? 8 : 0;
  const int kbb = hi ? 16 : 0;

  // Q fragments for this 16-row tile (16x128 bf16, kept resident)
  v16bf qf[4];
  const bf16_t* qrow = q_bf + (long)(tok0 + n15) * HIDDEN + h * DHEAD;
#pragma unroll
  for (int c2 = 0; c2 < 4; ++c2)
    qf[c2] = ld16(qrow + c2 * 32 + ka, qrow + c2 * 32 + ka + 16);

  v8f acc[8] = {};
  float m_run[8], l_run[8];
#pragma unroll
  for (int v = 0; v < 8; ++v) { m_run[v] = -3.0e38f; l_run[v] = 0.0f; }

  int kstart = qbase - W + 1; if (kstart < 0) kstart = 0; kstart &= ~31;
  const bf16_t* krow0 = k_bf + (long)b * S * (KV_HEADS * DHEAD) + kvh * DHEAD;
  const bf16_t* vt0   = vt_bf + (long)(b * KV_HEADS + kvh) * DHEAD * S;  // [d][s]
  bf16_t* myp = pl[wave];

  for (int kb = kstart; kb <= qbase + 15; kb += 32) {
    // ---- scores: two 16x16 tiles over 32 keys ----
    v8f sc[2] = {};
#pragma unroll
    for (int j = 0; j < 2; ++j) {
      const int key = kb + j * 16 + n15;
      const bf16_t* kp = krow0 + (long)key * (KV_HEADS * DHEAD) + ka;
      // Load all 4 K fragments first -> a single loadcnt wait, 4 WMMAs back-to-back.
      v16bf kf[4];
#pragma unroll
      for (int c2 = 0; c2 < 4; ++c2)
        kf[c2] = ld16(kp + c2 * 32, kp + c2 * 32 + 16);
#pragma unroll
      for (int c2 = 0; c2 < 4; ++c2)
        sc[j] = __builtin_amdgcn_wmma_f32_16x16x32_bf16(
            false, qf[c2], false, kf[c2], (short)0, sc[j], false, false);
      const int col = kb + j * 16 + n15;
#pragma unroll
      for (int v = 0; v < 8; ++v) {
        const int r = qbase + v + hi * 8;
        const int d = r - col;
        sc[j][v] = (d >= 0 && d < W) ? sc[j][v] * scale : -1.0e9f;
      }
    }
    // ---- online softmax (rows live inside one 16-lane half) ----
    float p0[8], p1[8], alpha[8];
#pragma unroll
    for (int v = 0; v < 8; ++v) {
      float mx = fmaxf(sc[0][v], sc[1][v]);
#pragma unroll
      for (int off = 1; off <= 8; off <<= 1) mx = fmaxf(mx, __shfl_xor(mx, off, 32));
      float mnew = fmaxf(m_run[v], mx);
      alpha[v] = __expf(m_run[v] - mnew);
      p0[v] = __expf(sc[0][v] - mnew);
      p1[v] = __expf(sc[1][v] - mnew);
      float ls = p0[v] + p1[v];
#pragma unroll
      for (int off = 1; off <= 8; off <<= 1) ls += __shfl_xor(ls, off, 32);
      l_run[v] = l_run[v] * alpha[v] + ls;
      m_run[v] = mnew;
    }
#pragma unroll
    for (int t = 0; t < 8; ++t)
#pragma unroll
      for (int v = 0; v < 8; ++v) acc[t][v] *= alpha[v];

    // ---- bounce P through LDS: C layout -> A layout ----
#pragma unroll
    for (int v = 0; v < 8; ++v) {
      const int r = v + hi * 8;
      myp[r * PLDS + n15]      = f2bf(p0[v]);
      myp[r * PLDS + 16 + n15] = f2bf(p1[v]);
    }
    asm volatile("s_wait_dscnt 0x0" ::: "memory");
    v16bf pf;
    {
      const bf16_t* pr = myp + n15 * PLDS + ka;
      BF16x16 u;
      u.h[0] = *(const v8bf*)(pr);
      u.h[1] = *(const v8bf*)(pr + 16);
      pf = u.v;
    }
    // ---- acc += P (16x32) x V (32x128), two groups of (4 loads -> 4 WMMAs) ----
#pragma unroll
    for (int g = 0; g < 2; ++g) {
      v16bf vf[4];
#pragma unroll
      for (int j2 = 0; j2 < 4; ++j2) {
        const bf16_t* vp = vt0 + (long)((g * 4 + j2) * 16 + n15) * S + kb + ka;
        vf[j2] = ld16(vp, vp + 16);
      }
#pragma unroll
      for (int j2 = 0; j2 < 4; ++j2)
        acc[g * 4 + j2] = __builtin_amdgcn_wmma_f32_16x16x32_bf16(
            false, pf, false, vf[j2], (short)0, acc[g * 4 + j2], false, false);
    }
  }

  // ---- normalize + emit bf16 attention output ----
  float inv_l[8];
#pragma unroll
  for (int v = 0; v < 8; ++v) inv_l[v] = 1.0f / l_run[v];
  bf16_t* orow = o_bf + (long)tok0 * HIDDEN + h * DHEAD;
#pragma unroll
  for (int j = 0; j < 8; ++j)
#pragma unroll
    for (int v = 0; v < 8; ++v) {
      const int r = v + hi * 8;
      orow[(long)r * HIDDEN + j * 16 + n15] = f2bf(acc[j][v] * inv_l[v]);
    }
}

// ---------------- host launcher ----------------
extern "C" void kernel_launch(void* const* d_in, const int* in_sizes, int n_in,
                              void* d_out, int out_size, void* d_ws, size_t ws_size,
                              hipStream_t stream) {
  const float* hidden = (const float*)d_in[0];
  const float* w_qkv  = (const float*)d_in[1];
  const float* b_qkv  = (const float*)d_in[2];
  const float* w_o    = (const float*)d_in[3];
  const float* b_o    = (const float*)d_in[4];
  const float* cosb   = (const float*)d_in[5];
  const float* sinb   = (const float*)d_in[6];
  const int*   seqp   = (const int*)d_in[7];
  const int*   winp   = (const int*)d_in[8];
  float* out = (float*)d_out;

  const long T = (long)in_sizes[0] / HIDDEN;   // 4096 (B*S)

  char* ws = (char*)d_ws;
  size_t off = 0;
  auto alloc = [&](size_t bytes) {
    char* p = ws + off;
    off = (off + bytes + 255) & ~(size_t)255;
    return p;
  };
  bf16_t* hbf    = (bf16_t*)alloc((size_t)T * HIDDEN * 2);
  bf16_t* wqkvbf = (bf16_t*)alloc((size_t)QKV_N * HIDDEN * 2);
  bf16_t* wobf   = (bf16_t*)alloc((size_t)HIDDEN * HIDDEN * 2);
  float*  qkv32  = (float*) alloc((size_t)T * QKV_N * 4);
  bf16_t* qbf    = (bf16_t*)alloc((size_t)T * HIDDEN * 2);
  bf16_t* kbf    = (bf16_t*)alloc((size_t)T * KV_HEADS * DHEAD * 2);
  bf16_t* vtbf   = (bf16_t*)alloc((size_t)T * KV_HEADS * DHEAD * 2);
  bf16_t* abf    = (bf16_t*)alloc((size_t)T * HIDDEN * 2);

  // 1) fp32 -> bf16 operand staging
  {
    long n4;
    n4 = (long)T * HIDDEN / 4;
    sc2_cvt_bf16<<<dim3((n4 + 255) / 256), dim3(256), 0, stream>>>(hidden, hbf, n4);
    n4 = (long)QKV_N * HIDDEN / 4;
    sc2_cvt_bf16<<<dim3((n4 + 255) / 256), dim3(256), 0, stream>>>(w_qkv, wqkvbf, n4);
    n4 = (long)HIDDEN * HIDDEN / 4;
    sc2_cvt_bf16<<<dim3((n4 + 255) / 256), dim3(256), 0, stream>>>(w_o, wobf, n4);
  }
  // 2) QKV projection (WMMA bf16, async LDS staging, fp32 accum + bias)
  sc2_gemm_bf16<<<dim3(QKV_N / 128, T / 64), dim3(GM_WAVES * 32), 0, stream>>>(
      hbf, wqkvbf, b_qkv, qkv32, QKV_N, HIDDEN);
  // 3) RoPE + repack (Q,K bf16; V transposed bf16)
  sc2_rope_pack<<<dim3((unsigned)T), dim3(256), 0, stream>>>(
      qkv32, cosb, sinb, qbf, kbf, vtbf, seqp);
  // 4) sliding-window GQA flash attention (WMMA)
  sc2_attn_wmma<<<dim3((unsigned)(T / (16 * ATT_WAVES)), H_HEADS),
                  dim3(ATT_WAVES * 32), 0, stream>>>(
      qbf, kbf, vtbf, abf, seqp, winp);
  // 5) output projection (WMMA bf16 + bias) -> fp32 out
  sc2_gemm_bf16<<<dim3(HIDDEN / 128, T / 64), dim3(GM_WAVES * 32), 0, stream>>>(
      abf, wobf, b_o, out, HIDDEN, HIDDEN);

  (void)n_in; (void)out_size; (void)ws_size;
}
